// GATE0a_50371376447988
// MI455X (gfx1250) — matrix operand order repaired and checked by reading, not compiled, not run on hardware
//
#include <hip/hip_runtime.h>

typedef __attribute__((ext_vector_type(16))) __bf16 v16bf;
typedef __attribute__((ext_vector_type(8)))  float  v8f;

union FragB { v16bf v; uint4 q[2]; };

#define NN_ 20000
#define NE_ 320000
#define NG_ 64

static __device__ __forceinline__ unsigned short f2bf_rne(float f) {
  unsigned int u = __float_as_uint(f);
  u += 0x7fffu + ((u >> 16) & 1u);          // round-to-nearest-even bf16
  return (unsigned short)(u >> 16);
}

// A-matrix described as up to 3 horizontally concatenated segments,
// each with optional row gather (x[row], x[col], edge_attr, ...).
struct SegDesc { const float* ptr; const int* gidx; int ld; int cols; };

// ---------------------------------------------------------------------------
// Fused gather-concat GEMM:  C[M x N] = act( A[M x K] * W[K x N] + b )
//   - no LDS: A fragments are 2 contiguous 8-float runs per lane (ISA 7.12.2
//     16-bit A layout), loaded as float4 pairs and packed to bf16 in regs.
//   - W pre-packed fragment-major: Wp[kc][n][kl] bf16, lane loads its 16
//     values (k = 8*g+(0..7), 16+8*g+(0..7)) as two b128 loads; L2-resident.
//   - 256 thr = 8 waves; block tile 128(M) x 64(N); wave = 16M x 64N,
//     4 WMMAs per 32-K chunk reusing one A fragment.
//   - NSEG compile-time: NSEG==1 requires Ktrue==Kpad (no guard at all);
//     NSEG==2 requires c0+c1==Ktrue==Kpad (single compare); NSEG==3 full
//     select with zero-fill for the K padding tail.
// ---------------------------------------------------------------------------
template <int NSEG>
__global__ __launch_bounds__(256)
void gemm_seg_wmma(SegDesc s0, SegDesc s1, SegDesc s2,
                   int M, int Ktrue, int Kpad, int N,
                   const unsigned short* __restrict__ Wb,
                   const float* __restrict__ bias,
                   float* __restrict__ Cc, int relu)
{
  const int tid  = threadIdx.x;
  const int lane = tid & 31;
  const int wv   = tid >> 5;          // 0..7 : which 16-row window
  const int l15  = lane & 15;
  const int gHi  = lane >> 4;
  const int bM   = blockIdx.y * 128;
  const int bN   = blockIdx.x * 64;

  const int c0  = s0.cols;
  const int c01 = c0 + s1.cols;

  // per-lane A row; gather indices hoisted out of the K loop
  const int mrow = bM + wv * 16 + l15;
  const int mc   = (mrow < M) ? mrow : (M - 1);
  const long r0 = s0.gidx ? (long)s0.gidx[mc] : (long)mc;
  const float* p0 = s0.ptr + r0 * s0.ld;
  const float* p1 = nullptr;
  const float* p2 = nullptr;
  if (NSEG >= 2) {
    const long r1 = s1.gidx ? (long)s1.gidx[mc] : (long)mc;
    p1 = s1.ptr + r1 * s1.ld;
  }
  if (NSEG >= 3) {
    const long r2 = s2.gidx ? (long)s2.gidx[mc] : (long)mc;
    p2 = s2.ptr + r2 * s2.ld;
  }

  v8f acc[4] = {};
  const uint4* wq = (const uint4*)Wb;

  for (int k0 = 0; k0 < Kpad; k0 += 32) {
    // ---- A fragment: a[0..7] <- k0+8g+(0..7), a[8..15] <- k0+16+8g+(0..7)
    float av[16];
    #pragma unroll
    for (int half = 0; half < 2; ++half) {
      const int kk = k0 + 16 * half + 8 * gHi;
      const float* src = nullptr; int c = 0;
      if (NSEG == 1) {
        src = p0; c = kk;                       // Ktrue == Kpad guaranteed
      } else if (NSEG == 2) {
        if (kk < c0) { src = p0; c = kk; }      // c0+c1 == Ktrue == Kpad
        else         { src = p1; c = kk - c0; }
      } else {
        if (kk < c0)         { src = p0; c = kk; }
        else if (kk < c01)   { src = p1; c = kk - c0; }
        else if (kk < Ktrue) { src = p2; c = kk - c01; }
      }
      float* o = av + half * 8;
      if (NSEG < 3 || src) {
        float4 qa = *(const float4*)(src + c);
        float4 qb = *(const float4*)(src + c + 4);
        o[0] = qa.x; o[1] = qa.y; o[2] = qa.z; o[3] = qa.w;
        o[4] = qb.x; o[5] = qb.y; o[6] = qb.z; o[7] = qb.w;
      } else {
        #pragma unroll
        for (int i = 0; i < 8; ++i) o[i] = 0.0f;
      }
    }
    v16bf a;
    #pragma unroll
    for (int h = 0; h < 16; ++h) a[h] = (__bf16)av[h];

    // ---- 4 B fragments from packed weights, 4 WMMAs (A reused) ----
    const long kb = ((long)(k0 >> 5) * N) << 2;     // uint4 base for chunk
    #pragma unroll
    for (int t = 0; t < 4; ++t) {
      const int n = bN + t * 16 + l15;
      const long u = kb + ((long)n << 2) + gHi;
      FragB fb;
      fb.q[0] = wq[u];
      fb.q[1] = wq[u + 2];
      acc[t] = __builtin_amdgcn_wmma_f32_16x16x32_bf16(
                 false, a, false, fb.v, (short)0, acc[t], false, false);
    }
  }

  // ---- epilogue: D layout -> VGPR r: M = r + 8*(lane>=16), N = lane&15 ----
  #pragma unroll
  for (int t = 0; t < 4; ++t) {
    const int n = bN + t * 16 + l15;
    const float bb = bias[n];
    #pragma unroll
    for (int r = 0; r < 8; ++r) {
      const int m = bM + wv * 16 + r + 8 * gHi;
      if (m < M) {
        float v = acc[t][r] + bb;
        if (relu) v = fmaxf(v, 0.0f);
        Cc[(long)m * N + n] = v;
      }
    }
  }
}

// ---------------------------------------------------------------------------
// Pack fp32 W[K x N] -> bf16 Wp[kc][n][kl] (kl = k%32), K zero-padded.
__global__ void pack_w(const float* __restrict__ W, int K, int N, int Kpad,
                       unsigned short* __restrict__ out)
{
  long tot = (long)Kpad * N;
  for (long i = (long)blockIdx.x * blockDim.x + threadIdx.x; i < tot;
       i += (long)gridDim.x * blockDim.x) {
    long kc  = i / ((long)N * 32);
    long rem = i - kc * (long)N * 32;
    int  n   = (int)(rem >> 5);
    int  kl  = (int)(rem & 31);
    int  k   = (int)(kc * 32 + kl);
    float v  = (k < K) ? W[(long)k * N + n] : 0.0f;
    out[i]   = f2bf_rne(v);
  }
}

__global__ void zero_f32(float* __restrict__ p, long n) {
  for (long i = (long)blockIdx.x * blockDim.x + threadIdx.x; i < n;
       i += (long)gridDim.x * blockDim.x) p[i] = 0.0f;
}

__global__ void scatter_count(const int* __restrict__ row, float* __restrict__ cnt) {
  int e = blockIdx.x * blockDim.x + threadIdx.x;
  if (e < NE_) atomicAdd(&cnt[row[e]], 1.0f);
}

__global__ void scatter_add_feat(const float* __restrict__ src, const int* __restrict__ row,
                                 int C, float* __restrict__ agg) {
  long tot = (long)NE_ * C;
  for (long i = (long)blockIdx.x * blockDim.x + threadIdx.x; i < tot;
       i += (long)gridDim.x * blockDim.x) {
    int e = (int)(i / C), c = (int)(i % C);
    atomicAdd(&agg[(long)row[e] * C + c], src[i]);
  }
}

__global__ void div_mean(float* __restrict__ agg, const float* __restrict__ cnt,
                         int n, int C) {
  long tot = (long)n * C;
  for (long i = (long)blockIdx.x * blockDim.x + threadIdx.x; i < tot;
       i += (long)gridDim.x * blockDim.x)
    agg[i] /= fmaxf(cnt[i / C], 1.0f);
}

// BN pass 1: per-column sum / sumsq (coalesced; partial per block + atomic)
__global__ void bn_pass1(const float* __restrict__ h, int Rows, int Cols,
                         float* __restrict__ gsum, float* __restrict__ gsq) {
  int c   = threadIdx.x % Cols;
  int rs  = threadIdx.x / Cols;
  int tpc = blockDim.x / Cols;            // 1 (C=256) or 2 (C=128)
  float s = 0.0f, s2 = 0.0f;
  for (long r = (long)blockIdx.x * tpc + rs; r < Rows; r += (long)gridDim.x * tpc) {
    float v = h[r * Cols + c]; s += v; s2 += v * v;
  }
  __shared__ float sh[256], sh2[256];
  sh[threadIdx.x] = s; sh2[threadIdx.x] = s2;
  __syncthreads();
  if (rs == 0) {
    for (int t = 1; t < tpc; ++t) { s += sh[t * Cols + c]; s2 += sh2[t * Cols + c]; }
    atomicAdd(&gsum[c], s); atomicAdd(&gsq[c], s2);
  }
}

// BN pass 2: normalize in place (biased variance, training-mode)
__global__ void bn_pass2(float* __restrict__ h, long total, int Rows, int Cols,
                         const float* __restrict__ gsum, const float* __restrict__ gsq,
                         const float* __restrict__ gamma, const float* __restrict__ beta) {
  for (long i = (long)blockIdx.x * blockDim.x + threadIdx.x; i < total;
       i += (long)gridDim.x * blockDim.x) {
    int c = (int)(i % Cols);
    float mu  = gsum[c] / (float)Rows;
    float var = gsq[c] / (float)Rows - mu * mu;
    float sc  = gamma[c] * rsqrtf(var + 1e-5f);
    h[i] = (h[i] - mu) * sc + beta[c];
  }
}

__global__ void pool_add(const float* __restrict__ x2, const int* __restrict__ batch,
                         float* __restrict__ pool) {
  long tot = (long)NN_ * 256;
  for (long i = (long)blockIdx.x * blockDim.x + threadIdx.x; i < tot;
       i += (long)gridDim.x * blockDim.x)
    atomicAdd(&pool[(long)batch[i >> 8] * 256 + (i & 255)], x2[i]);
}

__global__ void fc2_kernel(const float* __restrict__ h, const float* __restrict__ w,
                           const float* __restrict__ b, float* __restrict__ out) {
  int g = blockIdx.x * blockDim.x + threadIdx.x;
  if (g < NG_) {
    float s = b[0];
    for (int k = 0; k < 64; ++k) s += h[g * 64 + k] * w[k];
    out[g] = s;
  }
}

// ---------------------------------------------------------------------------
extern "C" void kernel_launch(void* const* d_in, const int* in_sizes, int n_in,
                              void* d_out, int out_size, void* d_ws, size_t ws_size,
                              hipStream_t stream)
{
  (void)in_sizes; (void)n_in; (void)out_size; (void)ws_size;
  const float* x     = (const float*)d_in[0];
  const int*   row   = (const int*)d_in[1];
  const int*   col   = row + NE_;
  const float* ea    = (const float*)d_in[2];
  const int*   batch = (const int*)d_in[3];

  const float* l1_eW1 = (const float*)d_in[4];  const float* l1_eb1 = (const float*)d_in[5];
  const float* l1_eW2 = (const float*)d_in[6];  const float* l1_eb2 = (const float*)d_in[7];
  const float* l1_n1W1= (const float*)d_in[8];  const float* l1_n1b1= (const float*)d_in[9];
  const float* l1_n1W2= (const float*)d_in[10]; const float* l1_n1b2= (const float*)d_in[11];
  const float* l1_n2W1= (const float*)d_in[12]; const float* l1_n2b1= (const float*)d_in[13];
  const float* l1_n2W2= (const float*)d_in[14]; const float* l1_n2b2= (const float*)d_in[15];
  const float* bn_n_g = (const float*)d_in[16]; const float* bn_n_b = (const float*)d_in[17];
  const float* bn_e_g = (const float*)d_in[18]; const float* bn_e_b = (const float*)d_in[19];
  const float* l2_eW1 = (const float*)d_in[20]; const float* l2_eb1 = (const float*)d_in[21];
  const float* l2_eW2 = (const float*)d_in[22]; const float* l2_eb2 = (const float*)d_in[23];
  const float* l2_n1W1= (const float*)d_in[24]; const float* l2_n1b1= (const float*)d_in[25];
  const float* l2_n1W2= (const float*)d_in[26]; const float* l2_n1b2= (const float*)d_in[27];
  const float* l2_n2W1= (const float*)d_in[28]; const float* l2_n2b1= (const float*)d_in[29];
  const float* l2_n2W2= (const float*)d_in[30]; const float* l2_n2b2= (const float*)d_in[31];
  const float* fc1W   = (const float*)d_in[32]; const float* fc1b   = (const float*)d_in[33];
  const float* fc2W   = (const float*)d_in[34]; const float* fc2b   = (const float*)d_in[35];

  char* wsp = (char*)d_ws;
  size_t off = 0;
  auto alloc = [&](size_t bytes) -> char* {
    char* p = wsp + off;
    off = (off + bytes + 255) & ~(size_t)255;
    return p;
  };
  float* T1  = (float*)alloc((size_t)NE_ * 256 * 4);
  float* T2  = (float*)alloc((size_t)NE_ * 256 * 4);
  float* E1  = (float*)alloc((size_t)NE_ * 128 * 4);
  float* AGG = (float*)alloc((size_t)NN_ * 256 * 4);
  float* CNT = (float*)alloc((size_t)NN_ * 4);
  float* X1  = (float*)alloc((size_t)NN_ * 256 * 4);
  float* XT  = (float*)alloc((size_t)NN_ * 256 * 4);
  float* X2  = (float*)alloc((size_t)NN_ * 256 * 4);
  float* POOL= (float*)alloc((size_t)NG_ * 256 * 4);
  float* H1  = (float*)alloc((size_t)NG_ * 64 * 4);
  float* RS  = (float*)alloc(256 * 4);
  float* RQ  = (float*)alloc(256 * 4);

  // --- weights -> packed bf16 fragments [kc][n][kl], K zero-padded to 32 ---
  const int kArr[13]    = {144, 64, 192, 128, 192, 128, 640, 128, 384, 256, 512, 256, 256};
  const int nArr[13]    = { 64,128, 128, 128, 128, 256, 128, 128, 256, 256, 256, 256,  64};
  const float* wF[13]   = { l1_eW1, l1_eW2, l1_n1W1, l1_n1W2, l1_n2W1, l1_n2W2,
                            l2_eW1, l2_eW2, l2_n1W1, l2_n1W2, l2_n2W1, l2_n2W2, fc1W };
  unsigned short* wb[13];
  int kp[13];
  for (int i = 0; i < 13; ++i) {
    kp[i] = (kArr[i] + 31) & ~31;
    wb[i] = (unsigned short*)alloc((size_t)kp[i] * nArr[i] * 2);
    long tot = (long)kp[i] * nArr[i];
    pack_w<<<dim3((unsigned)((tot + 255) / 256)), dim3(256), 0, stream>>>(
        wF[i], kArr[i], nArr[i], kp[i], wb[i]);
  }

  SegDesc Z{nullptr, nullptr, 0, 0};
  auto gemm = [&](SegDesc a0, SegDesc a1, SegDesc a2, int nseg,
                  int M, int K, int Kp, int N,
                  const unsigned short* Wb_, const float* bias, float* Cc, int relu) {
    dim3 grid((unsigned)(N / 64), (unsigned)((M + 127) / 128));
    if (nseg == 1)
      gemm_seg_wmma<1><<<grid, dim3(256), 0, stream>>>(a0, a1, a2, M, K, Kp, N, Wb_, bias, Cc, relu);
    else if (nseg == 2)
      gemm_seg_wmma<2><<<grid, dim3(256), 0, stream>>>(a0, a1, a2, M, K, Kp, N, Wb_, bias, Cc, relu);
    else
      gemm_seg_wmma<3><<<grid, dim3(256), 0, stream>>>(a0, a1, a2, M, K, Kp, N, Wb_, bias, Cc, relu);
  };
  auto zero = [&](float* p, long n) {
    zero_f32<<<dim3(2048), dim3(256), 0, stream>>>(p, n);
  };

  // ===================== Layer 1 =====================
  // eh = relu([x[row] | x[col] | ea] @ l1_eW1 + b)     (320k x 64)
  gemm({x,row,64,64}, {x,col,64,64}, {ea,nullptr,16,16}, 3, NE_, 144, 160, 64, wb[0], l1_eb1, T1, 1);
  // e1 = eh @ l1_eW2 + b                               (320k x 128)
  gemm({T1,nullptr,64,64}, Z, Z, 1, NE_, 64, 64, 128, wb[1], l1_eb2, E1, 0);
  // mh = relu([x[col] | e1] @ l1_n1W1 + b)             (320k x 128)
  gemm({x,col,64,64}, {E1,nullptr,128,128}, Z, 2, NE_, 192, 192, 128, wb[2], l1_n1b1, T1, 1);
  // m = mh @ l1_n1W2 + b                               (320k x 128)
  gemm({T1,nullptr,128,128}, Z, Z, 1, NE_, 128, 128, 128, wb[3], l1_n1b2, T2, 0);
  // agg = scatter_mean(m, row)
  zero(AGG, (long)NN_ * 256); zero(CNT, NN_);
  scatter_count<<<dim3((NE_ + 255) / 256), dim3(256), 0, stream>>>(row, CNT);
  scatter_add_feat<<<dim3(4096), dim3(256), 0, stream>>>(T2, row, 128, AGG);
  div_mean<<<dim3(2048), dim3(256), 0, stream>>>(AGG, CNT, NN_, 128);
  // x1 = ([x | agg] @ n2W1 relu) @ n2W2 + b            (20k x 256)
  gemm({x,nullptr,64,64}, {AGG,nullptr,128,128}, Z, 2, NN_, 192, 192, 128, wb[4], l1_n2b1, XT, 1);
  gemm({XT,nullptr,128,128}, Z, Z, 1, NN_, 128, 128, 256, wb[5], l1_n2b2, X1, 0);

  // BatchNorm (training-mode batch stats), in place
  zero(RS, 256); zero(RQ, 256);
  bn_pass1<<<dim3(512), dim3(256), 0, stream>>>(X1, NN_, 256, RS, RQ);
  bn_pass2<<<dim3(2048), dim3(256), 0, stream>>>(X1, (long)NN_ * 256, NN_, 256, RS, RQ, bn_n_g, bn_n_b);
  zero(RS, 256); zero(RQ, 256);
  bn_pass1<<<dim3(512), dim3(256), 0, stream>>>(E1, NE_, 128, RS, RQ);
  bn_pass2<<<dim3(4096), dim3(256), 0, stream>>>(E1, (long)NE_ * 128, NE_, 128, RS, RQ, bn_e_g, bn_e_b);

  // ===================== Layer 2 =====================
  // eh2 = relu([x1[row] | x1[col] | e1] @ l2_eW1 + b)  (320k x 128), K=640
  gemm({X1,row,256,256}, {X1,col,256,256}, {E1,nullptr,128,128}, 3, NE_, 640, 640, 128, wb[6], l2_eb1, T1, 1);
  gemm({T1,nullptr,128,128}, Z, Z, 1, NE_, 128, 128, 128, wb[7], l2_eb2, T2, 0);          // e2
  gemm({X1,col,256,256}, {T2,nullptr,128,128}, Z, 2, NE_, 384, 384, 256, wb[8], l2_n1b1, T1, 1); // mh2
  gemm({T1,nullptr,256,256}, Z, Z, 1, NE_, 256, 256, 256, wb[9], l2_n1b2, T2, 0);         // m2
  zero(AGG, (long)NN_ * 256); zero(CNT, NN_);
  scatter_count<<<dim3((NE_ + 255) / 256), dim3(256), 0, stream>>>(row, CNT);
  scatter_add_feat<<<dim3(4096), dim3(256), 0, stream>>>(T2, row, 256, AGG);
  div_mean<<<dim3(2048), dim3(256), 0, stream>>>(AGG, CNT, NN_, 256);
  gemm({X1,nullptr,256,256}, {AGG,nullptr,256,256}, Z, 2, NN_, 512, 512, 256, wb[10], l2_n2b1, XT, 1);
  gemm({XT,nullptr,256,256}, Z, Z, 1, NN_, 256, 256, 256, wb[11], l2_n2b2, X2, 0);        // x2

  // ===================== Pool + readout =====================
  zero(POOL, (long)NG_ * 256);
  pool_add<<<dim3(2048), dim3(256), 0, stream>>>(X2, batch, POOL);
  gemm({POOL,nullptr,256,256}, Z, Z, 1, NG_, 256, 256, 64, wb[12], fc1b, H1, 1);          // fc1+relu
  fc2_kernel<<<dim3(1), dim3(64), 0, stream>>>(H1, fc2W, fc2b, (float*)d_out);            // fc2
}
